// MultiHeadAttention_10101763080672
// MI455X (gfx1250) — matrix operand order repaired and checked by reading, not compiled, not run on hardware
//
#include <hip/hip_runtime.h>

typedef __bf16 bf16_t;
typedef __attribute__((ext_vector_type(16))) __bf16 v16bf;
typedef __attribute__((ext_vector_type(8)))  float   v8f;
typedef __attribute__((ext_vector_type(4)))  unsigned int u32x4;

#define BB 2
#define NN 4096
#define CC 512
#define HH 8
#define DH 64

__device__ __forceinline__ bf16_t to_bf16(float f) {
  unsigned u = __builtin_bit_cast(unsigned, f);
  unsigned r = (u + 0x7FFFu + ((u >> 16) & 1u)) >> 16;  // RNE
  unsigned short s = (unsigned short)r;
  return __builtin_bit_cast(bf16_t, s);
}

__device__ __forceinline__ v8f wmma_bf16(v16bf a, v16bf b, v8f c) {
  return __builtin_amdgcn_wmma_f32_16x16x32_bf16(false, a, false, b, (short)0, c,
                                                 false, false);
}

// A-fragment (16x32 bf16), row-major source, ld in elements.
// lanes 0-15: row m, K = k0+0..7 (v0-3) and k0+16..23 (v4-7);
// lanes 16-31: K = k0+8..15 and k0+24..31.
__device__ __forceinline__ v16bf load_a_frag(const bf16_t* base, int ld,
                                             int row0, int k0) {
  int lane = threadIdx.x & 31;
  int m = lane & 15;
  int half = lane >> 4;
  const bf16_t* p = base + (size_t)(row0 + m) * ld + k0 + half * 8;
  union { u32x4 q[2]; v16bf v; } u;
  u.q[0] = *(const u32x4*)p;         // K chunk +0
  u.q[1] = *(const u32x4*)(p + 16);  // K chunk +16
  return u.v;
}

// B-fragment (32x16 bf16) from the TRANSPOSED operand BT (row-major [n][k]).
// lane col n = lane%16; lanes 0-15 hold K=k0+0..7,+16..23; lanes 16-31 K=k0+8..15,+24..31.
__device__ __forceinline__ v16bf load_b_frag(const bf16_t* bt, int ld,
                                             int n0, int k0) {
  int lane = threadIdx.x & 31;
  int n = lane & 15;
  int half = lane >> 4;
  const bf16_t* p = bt + (size_t)(n0 + n) * ld + k0 + half * 8;
  union { u32x4 q[2]; v16bf v; } u;
  u.q[0] = *(const u32x4*)p;
  u.q[1] = *(const u32x4*)(p + 16);
  return u.v;
}

// ---------------- prep kernels ----------------
__global__ void cvt_x_kernel(const float* __restrict__ x, bf16_t* __restrict__ xb, int n) {
  int i = blockIdx.x * blockDim.x + threadIdx.x;
  if (i < n) xb[i] = to_bf16(x[i]);
}

// w [rows][cols] f32 -> wT [cols][rows] bf16
__global__ void transpose_w_kernel(const float* __restrict__ w, bf16_t* __restrict__ wT,
                                   int rows, int cols) {
  int i = blockIdx.x * blockDim.x + threadIdx.x;
  if (i < rows * cols) {
    int r = i / cols, c = i - r * cols;
    wT[(size_t)c * rows + r] = to_bf16(w[i]);
  }
}

// ---------------- QKV projection ----------------
// grid (24, 64), block 128. Wave computes a 32x64 tile of [8192 x 1536]:
// 2 A-frags + 4 B-frags -> 8 wmma per K-step (2x wmma per B-load vs 16-row tile).
__global__ __launch_bounds__(128) void qkv_kernel(
    const bf16_t* __restrict__ xb, const bf16_t* __restrict__ wT,
    const float* __restrict__ bias,
    bf16_t* __restrict__ q, bf16_t* __restrict__ k, bf16_t* __restrict__ vt) {
  int wave = threadIdx.x >> 5;
  int lane = threadIdx.x & 31;
  int row0 = (blockIdx.y * 4 + wave) * 32;
  int col0 = blockIdx.x * 64;

  v8f acc[2][4] = {};
  for (int k0 = 0; k0 < CC; k0 += 32) {
    v16bf a0 = load_a_frag(xb, CC, row0, k0);
    v16bf a1 = load_a_frag(xb, CC, row0 + 16, k0);
#pragma unroll
    for (int j = 0; j < 4; ++j) {
      v16bf b = load_b_frag(wT, CC, col0 + j * 16, k0);
      acc[0][j] = wmma_bf16(a0, b, acc[0][j]);
      acc[1][j] = wmma_bf16(a1, b, acc[1][j]);
    }
  }

  const float scale = 0.04419417382415922f;  // 512^-0.5 (reference scales by C)
  int colL = lane & 15, half = lane >> 4;
#pragma unroll
  for (int i = 0; i < 2; ++i) {
#pragma unroll
    for (int j = 0; j < 4; ++j) {
      int col = col0 + j * 16 + colL;
      float bv = bias[col];
      int which = col >> 9;       // 0=q 1=k 2=v
      int cc = col & (CC - 1);
      int h = cc >> 6;
      int dh = cc & (DH - 1);
#pragma unroll
      for (int r = 0; r < 8; ++r) {
        int row = row0 + i * 16 + half * 8 + r;
        int b_ = row >> 12;               // /4096
        int n = row & (NN - 1);
        float val = acc[i][j][r] + bv;
        size_t bh = (size_t)(b_ * HH + h);
        if (which == 0)
          q[(bh * NN + n) * DH + dh] = to_bf16(val * scale);
        else if (which == 1)
          k[(bh * NN + n) * DH + dh] = to_bf16(val);
        else
          vt[(bh * DH + dh) * NN + n] = to_bf16(val);
      }
    }
  }
}

// ---------------- fused flash attention ----------------
// grid (64, 16), block 128. Wave owns 16 query rows; keys in blocks of 32.
// V fragments are prefetched at the top of the loop so their global loads
// overlap the softmax VALU work and the LDS re-swizzle (fence below would
// otherwise pin them).
__global__ __launch_bounds__(128) void attn_kernel(
    const bf16_t* __restrict__ qg, const bf16_t* __restrict__ kg,
    const bf16_t* __restrict__ vtg, bf16_t* __restrict__ og) {
  __shared__ __align__(16) bf16_t pst[4][16][48];  // per-wave P staging, padded

  int wave = threadIdx.x >> 5;
  int lane = threadIdx.x & 31;
  int bh = blockIdx.y;
  int q0 = blockIdx.x * 64 + wave * 16;

  const bf16_t* qb = qg + (size_t)bh * NN * DH;
  const bf16_t* kb = kg + (size_t)bh * NN * DH;
  const bf16_t* vb = vtg + (size_t)bh * DH * NN;

  v16bf qa0 = load_a_frag(qb, DH, q0, 0);
  v16bf qa1 = load_a_frag(qb, DH, q0, 32);

  float m[8], l[8];
#pragma unroll
  for (int r = 0; r < 8; ++r) { m[r] = -3.0e38f; l[r] = 0.f; }
  v8f of[4] = {};

  int colL = lane & 15, half = lane >> 4;

  for (int kb0 = 0; kb0 < NN; kb0 += 32) {
    // ---- prefetch V fragments for this key block (used only at the end) ----
    v16bf vf[4];
#pragma unroll
    for (int j = 0; j < 4; ++j) vf[j] = load_b_frag(vb, NN, j * 16, kb0);

    // ---- S tile: 16 rows x 32 keys (2 fragments), K-loop over DH ----
    v8f s0 = {}, s1 = {};
    {
      v16bf b00 = load_b_frag(kb, DH, kb0, 0);
      v16bf b01 = load_b_frag(kb, DH, kb0 + 16, 0);
      s0 = wmma_bf16(qa0, b00, s0);
      s1 = wmma_bf16(qa0, b01, s1);
      v16bf b10 = load_b_frag(kb, DH, kb0, 32);
      v16bf b11 = load_b_frag(kb, DH, kb0 + 16, 32);
      s0 = wmma_bf16(qa1, b10, s0);
      s1 = wmma_bf16(qa1, b11, s1);
    }

    // ---- online softmax (rows live across lanes 0-15 / 16-31) ----
    float mx[8];
#pragma unroll
    for (int r = 0; r < 8; ++r) mx[r] = fmaxf(s0[r], s1[r]);
#pragma unroll
    for (int d = 1; d < 16; d <<= 1)
#pragma unroll
      for (int r = 0; r < 8; ++r) mx[r] = fmaxf(mx[r], __shfl_xor(mx[r], d, 32));

    float alpha[8], rs[8];
#pragma unroll
    for (int r = 0; r < 8; ++r) {
      float mn = fmaxf(m[r], mx[r]);
      alpha[r] = __expf(m[r] - mn);
      m[r] = mn;
      float p0 = __expf(s0[r] - mn);
      float p1 = __expf(s1[r] - mn);
      s0[r] = p0; s1[r] = p1;
      rs[r] = p0 + p1;
    }
#pragma unroll
    for (int d = 1; d < 16; d <<= 1)
#pragma unroll
      for (int r = 0; r < 8; ++r) rs[r] += __shfl_xor(rs[r], d, 32);
#pragma unroll
    for (int r = 0; r < 8; ++r) l[r] = l[r] * alpha[r] + rs[r];
#pragma unroll
    for (int j = 0; j < 4; ++j)
#pragma unroll
      for (int r = 0; r < 8; ++r) of[j][r] *= alpha[r];

    // ---- re-swizzle P: C-layout -> A-layout via LDS ----
#pragma unroll
    for (int r = 0; r < 8; ++r) {
      pst[wave][half * 8 + r][colL]      = to_bf16(s0[r]);
      pst[wave][half * 8 + r][16 + colL] = to_bf16(s1[r]);
    }
    asm volatile("s_wait_dscnt 0" ::: "memory");  // same-wave LDS RAW fence
    v16bf pa = load_a_frag(&pst[wave][0][0], 48, 0, 0);

    // ---- PV: O += P(16x32) x V(32x64), V already in registers ----
#pragma unroll
    for (int j = 0; j < 4; ++j) of[j] = wmma_bf16(pa, vf[j], of[j]);

    asm volatile("s_wait_dscnt 0" ::: "memory");  // WAR before next P store
  }

  // epilogue: O[row][dh] / l -> [B,N,C] bf16
  int b_ = bh >> 3, h = bh & 7;
#pragma unroll
  for (int r = 0; r < 8; ++r) l[r] = 1.0f / l[r];
#pragma unroll
  for (int j = 0; j < 4; ++j)
#pragma unroll
    for (int r = 0; r < 8; ++r) {
      int row = q0 + half * 8 + r;
      og[((size_t)(b_ * NN + row)) * CC + h * DH + j * 16 + colL] =
          to_bf16(of[j][r] * l[r]);
    }
}

// ---------------- output projection ----------------
// grid (8, 64), block 128. Wave computes 32x64; out = O @ w_out + b_out (f32).
__global__ __launch_bounds__(128) void proj_kernel(
    const bf16_t* __restrict__ ob, const bf16_t* __restrict__ wT,
    const float* __restrict__ bias, float* __restrict__ out) {
  int wave = threadIdx.x >> 5;
  int lane = threadIdx.x & 31;
  int row0 = (blockIdx.y * 4 + wave) * 32;
  int col0 = blockIdx.x * 64;

  v8f acc[2][4] = {};
  for (int k0 = 0; k0 < CC; k0 += 32) {
    v16bf a0 = load_a_frag(ob, CC, row0, k0);
    v16bf a1 = load_a_frag(ob, CC, row0 + 16, k0);
#pragma unroll
    for (int j = 0; j < 4; ++j) {
      v16bf b = load_b_frag(wT, CC, col0 + j * 16, k0);
      acc[0][j] = wmma_bf16(a0, b, acc[0][j]);
      acc[1][j] = wmma_bf16(a1, b, acc[1][j]);
    }
  }
  int colL = lane & 15, half = lane >> 4;
#pragma unroll
  for (int i = 0; i < 2; ++i)
#pragma unroll
    for (int j = 0; j < 4; ++j) {
      int col = col0 + j * 16 + colL;
      float bv = bias[col];
#pragma unroll
      for (int r = 0; r < 8; ++r) {
        int row = row0 + i * 16 + half * 8 + r;
        out[(size_t)row * CC + col] = acc[i][j][r] + bv;
      }
    }
}

extern "C" void kernel_launch(void* const* d_in, const int* in_sizes, int n_in,
                              void* d_out, int out_size, void* d_ws, size_t ws_size,
                              hipStream_t stream) {
  const float* x     = (const float*)d_in[0];  // [B,N,C]
  const float* w_qkv = (const float*)d_in[1];  // [C,3C]
  const float* b_qkv = (const float*)d_in[2];  // [3C]
  const float* w_out = (const float*)d_in[3];  // [C,C]
  const float* b_out = (const float*)d_in[4];  // [C]
  float* out = (float*)d_out;

  char* ws = (char*)d_ws;
  const size_t ROWS = (size_t)BB * NN;                 // 8192
  bf16_t* xb    = (bf16_t*)ws;  ws += ROWS * CC * 2;            // 8 MB
  bf16_t* wqkvT = (bf16_t*)ws;  ws += (size_t)3 * CC * CC * 2;  // 1.5 MB
  bf16_t* woutT = (bf16_t*)ws;  ws += (size_t)CC * CC * 2;      // 0.5 MB
  bf16_t* qbuf  = (bf16_t*)ws;  ws += (size_t)BB * HH * NN * DH * 2;  // 8 MB
  bf16_t* kbuf  = (bf16_t*)ws;  ws += (size_t)BB * HH * NN * DH * 2;  // 8 MB
  bf16_t* vtbuf = (bf16_t*)ws;  ws += (size_t)BB * HH * DH * NN * 2;  // 8 MB
  bf16_t* obuf  = (bf16_t*)ws;  ws += ROWS * CC * 2;                  // 8 MB

  int nx = (int)(ROWS * CC);
  cvt_x_kernel<<<(nx + 255) / 256, 256, 0, stream>>>(x, xb, nx);
  transpose_w_kernel<<<(CC * 3 * CC + 255) / 256, 256, 0, stream>>>(w_qkv, wqkvT, CC, 3 * CC);
  transpose_w_kernel<<<(CC * CC + 255) / 256, 256, 0, stream>>>(w_out, woutT, CC, CC);

  qkv_kernel<<<dim3(24, 64), 128, 0, stream>>>(xb, wqkvT, b_qkv, qbuf, kbuf, vtbuf);
  attn_kernel<<<dim3(64, 16), 128, 0, stream>>>(qbuf, kbuf, vtbuf, obuf);
  proj_kernel<<<dim3(8, 64), 128, 0, stream>>>(obuf, woutT, b_out, out);
}
